// MMoE_68813966016791
// MI455X (gfx1250) — matrix operand order repaired and checked by reading, not compiled, not run on hardware
//
#include <hip/hip_runtime.h>

// ---------------------------------------------------------------------------
// MMoE forward on gfx1250 (MI455X): f16 WMMA GEMMs with async global->LDS
// double-buffered staging + fused epilogues.
// ---------------------------------------------------------------------------

typedef __attribute__((ext_vector_type(16))) _Float16 v16h;
typedef __attribute__((ext_vector_type(8)))  _Float16 v8h;
typedef __attribute__((ext_vector_type(8)))  float    v8f;

static constexpr int Bsz = 16384;  // batch
static constexpr int Dd  = 768;    // input dim
static constexpr int H1  = 256;    // expert hidden
static constexpr int H2  = 128;    // expert out
static constexpr int GH  = 128;    // gate hidden
static constexpr int NE  = 12;     // experts
static constexpr int NT  = 12;     // tasks

// -------------------------- CDNA5 async helpers ----------------------------

// GLOBAL_LOAD_ASYNC_TO_LDS_B128: VDST = VGPR holding LDS byte address,
// VADDR = 64-bit global address. Tracked with ASYNCcnt (no VGPR data path).
__device__ __forceinline__ void async_copy_b128(const void* gptr, void* lptr) {
    // low 32 bits of a generic pointer into LDS == LDS byte address
    unsigned lds_addr = (unsigned)(unsigned long long)lptr;
    asm volatile("global_load_async_to_lds_b128 %0, %1, off"
                 :: "v"(lds_addr), "v"(gptr) : "memory");
}

__device__ __forceinline__ void wait_async0() {
    asm volatile("s_wait_asynccnt 0x0" ::: "memory");
}

// -------------------------- conversion kernels -----------------------------

__global__ __launch_bounds__(256) void cvt_f32_to_f16(const float* __restrict__ src,
                                                      _Float16* __restrict__ dst,
                                                      size_t n) {
    size_t i = (size_t)blockIdx.x * blockDim.x + threadIdx.x;
    size_t stride = (size_t)gridDim.x * blockDim.x;
    for (; i < n; i += stride) dst[i] = (_Float16)src[i];
}

// src: (batch, K, N) f32 row-major  ->  dst: (batch, N, K) f16 row-major
__global__ __launch_bounds__(256) void transpose_cvt(const float* __restrict__ src,
                                                     _Float16* __restrict__ dst,
                                                     int batch, int K, int N) {
    size_t total = (size_t)batch * K * N;
    size_t i = (size_t)blockIdx.x * blockDim.x + threadIdx.x;
    size_t stride = (size_t)gridDim.x * blockDim.x;
    for (; i < total; i += stride) {
        size_t z = i / ((size_t)K * N);
        size_t r = i % ((size_t)K * N);
        int n = (int)(r / K);
        int k = (int)(r % K);
        dst[i] = (_Float16)src[z * (size_t)K * N + (size_t)k * N + n];
    }
}

// ------------------------ WMMA GEMM + bias + ReLU --------------------------
// C(MxN) = relu(A(MxK) * B(KxN) + bias), B given transposed as BT(NxK).
// Block tile: 128(M) x 64(N), K-step 32. 4 waves; wave w owns rows
// [w*32, w*32+32) => 8 accumulators/wave, 8 WMMA per K-step.
// Double-buffered LDS filled with async global->LDS copies.

#define TM 128
#define TN 64
#define TK 32
#define LDSW 40   // padded row stride (halves); 80B rows keep 16B alignment

#define BUF_HALVES ((TM + TN) * LDSW)   // one buffer: A then B

__device__ __forceinline__ v16h frag16(const _Float16* p, int o0, int o1) {
    v16h f;
    ((v8h*)&f)[0] = *(const v8h*)(p + o0);
    ((v8h*)&f)[1] = *(const v8h*)(p + o1);
    return f;
}

__global__ __launch_bounds__(128) void gemm_relu_f16(
    const _Float16* __restrict__ A, int lda,
    const _Float16* __restrict__ BT,     // (N x K) row-major
    const float* __restrict__ bias,      // (N)
    _Float16* __restrict__ C,            // (M x N) row-major, relu'd f16
    int N, int K) {
    __shared__ __align__(16) _Float16 lds[2 * BUF_HALVES];

    const int tid   = threadIdx.x;
    const int wave  = tid >> 5;
    const int lane  = tid & 31;
    const int l16   = lane & 15;
    const int khalf = lane >> 4;          // 0 or 1
    const int m0 = blockIdx.x * TM;
    const int n0 = blockIdx.y * TN;

    // async staging map: per K-tile, A = 128x32 halves (512 x 16B chunks),
    // B = 64x32 halves (256 x 16B chunks); 128 threads x (2 A + 1 B) copies.
    const int srow = tid >> 1;            // 0..63
    const int soff = (tid & 1) * 16;      // 0 or 16 halves

    v8f acc[2][4] = {};

    const int nk = K / TK;

    auto stage = [&](int s, int k0) {
        const _Float16* ag0 = A  + (size_t)(m0 + srow) * lda + k0 + soff;
        const _Float16* ag1 = A  + (size_t)(m0 + 64 + srow) * lda + k0 + soff;
        const _Float16* bg  = BT + (size_t)(n0 + srow) * K   + k0 + soff;
        _Float16* base = &lds[s * BUF_HALVES];
        async_copy_b128(ag0, base + srow * LDSW + soff);
        async_copy_b128(ag1, base + (64 + srow) * LDSW + soff);
        async_copy_b128(bg,  base + TM * LDSW + srow * LDSW + soff);
    };

    stage(0, 0);

    for (int i = 0; i < nk; ++i) {
        wait_async0();       // this wave's copies into buf[i&1] complete
        __syncthreads();     // all waves' copies complete; prior compute done
        if (i + 1 < nk) stage((i + 1) & 1, (i + 1) * TK);

        const _Float16* Abase = &lds[(i & 1) * BUF_HALVES];
        const _Float16* Bbase = Abase + TM * LDSW;

        // A fragments (16x32): lanes 0-15 K={0..7,16..23}; lanes 16-31 K={8..15,24..31}
        const _Float16* ap = Abase + (wave * 32 + l16) * LDSW;
        v16h a0 = frag16(ap,             khalf * 8, 16 + khalf * 8);
        v16h a1 = frag16(ap + 16 * LDSW, khalf * 8, 16 + khalf * 8);

        #pragma unroll
        for (int jn = 0; jn < 4; ++jn) {
            // B fragment (32x16): lanes 0-15 K={0..7,16..23}; lanes 16-31 K={8..15,24..31}
            const _Float16* bp = Bbase + (jn * 16 + l16) * LDSW;
            v16h b = frag16(bp, khalf * 8, 16 + khalf * 8);
            acc[0][jn] = __builtin_amdgcn_wmma_f32_16x16x32_f16(
                false, a0, false, b, (short)0, acc[0][jn], false, false);
            acc[1][jn] = __builtin_amdgcn_wmma_f32_16x16x32_f16(
                false, a1, false, b, (short)0, acc[1][jn], false, false);
        }
        __syncthreads();
    }

    // Epilogue: C/D layout => column = l16 (per 16-wide subtile),
    // row = r + 8*khalf within each 16-row subtile.
    #pragma unroll
    for (int jm = 0; jm < 2; ++jm) {
        const int rowBase = m0 + wave * 32 + jm * 16 + 8 * khalf;
        #pragma unroll
        for (int jn = 0; jn < 4; ++jn) {
            const int col = n0 + jn * 16 + l16;
            const float bj = bias[col];
            #pragma unroll
            for (int r = 0; r < 8; ++r) {
                float v = acc[jm][jn][r] + bj;
                v = fmaxf(v, 0.0f);
                C[(size_t)(rowBase + r) * N + col] = (_Float16)v;
            }
        }
    }
}

// -------------------- gate layer 2 + softmax over experts ------------------
// g: (B, GH) f16 for one task; Wg2: (GH, NE) f32; gates: (B, NE) f32
__global__ __launch_bounds__(256) void gate2_softmax_kernel(
    const _Float16* __restrict__ g, const float* __restrict__ Wg2,
    const float* __restrict__ bg2, float* __restrict__ gates) {
    const int b = blockIdx.x * 256 + threadIdx.x;
    float acc[NE];
    #pragma unroll
    for (int e = 0; e < NE; ++e) acc[e] = bg2[e];
    const _Float16* gr = g + (size_t)b * GH;
    for (int k = 0; k < GH; ++k) {
        const float xv = (float)gr[k];
        const float* wr = Wg2 + (size_t)k * NE;
        #pragma unroll
        for (int e = 0; e < NE; ++e) acc[e] += xv * wr[e];
    }
    float mx = acc[0];
    #pragma unroll
    for (int e = 1; e < NE; ++e) mx = fmaxf(mx, acc[e]);
    float s = 0.0f;
    #pragma unroll
    for (int e = 0; e < NE; ++e) { acc[e] = __expf(acc[e] - mx); s += acc[e]; }
    const float inv = 1.0f / s;
    float* outp = gates + (size_t)b * NE;
    #pragma unroll
    for (int e = 0; e < NE; ++e) outp[e] = acc[e] * inv;
}

// -------------- combine (gate-weighted experts) + per-task head ------------

struct HeadParams {
    const float* Wc[NT];   // (H2, n_t)
    const float* bc[NT];   // (n_t)
    float*       out[NT];  // (B, n_t)
    int          n[NT];
};

__global__ __launch_bounds__(128) void combine_head_kernel(
    const _Float16* __restrict__ experts,  // (NE, B, H2) f16
    const float* __restrict__ gates,       // (NT, B, NE) f32
    HeadParams hp) {
    const int b = blockIdx.x;
    const int t = blockIdx.y;
    const int k = threadIdx.x;  // 0..127

    __shared__ float comb[H2];
    __shared__ float logits[16];

    const float* gp = gates + ((size_t)t * Bsz + b) * NE;
    float c = 0.0f;
    #pragma unroll
    for (int e = 0; e < NE; ++e)
        c += gp[e] * (float)experts[((size_t)e * Bsz + b) * H2 + k];
    comb[k] = c;
    __syncthreads();

    const int nt = hp.n[t];
    if (k < nt) {
        const float* W = hp.Wc[t];
        float acc = hp.bc[t][k];
        for (int kk = 0; kk < H2; ++kk) acc += comb[kk] * W[(size_t)kk * nt + k];
        logits[k] = acc;
    }
    __syncthreads();
    if (k < nt) {
        float mx = -3.0e38f;
        for (int i = 0; i < nt; ++i) mx = fmaxf(mx, logits[i]);
        float s = 0.0f;
        for (int i = 0; i < nt; ++i) s += __expf(logits[i] - mx);
        hp.out[t][(size_t)b * nt + k] = __expf(logits[k] - mx) / s;
    }
}

// ------------------------------- launcher ----------------------------------

extern "C" void kernel_launch(void* const* d_in, const int* in_sizes, int n_in,
                              void* d_out, int out_size, void* d_ws, size_t ws_size,
                              hipStream_t stream) {
    (void)in_sizes; (void)n_in; (void)out_size; (void)ws_size;

    const float* x    = (const float*)d_in[0];
    const float* We1  = (const float*)d_in[1];
    const float* be1  = (const float*)d_in[2];
    const float* We2  = (const float*)d_in[3];
    const float* be2  = (const float*)d_in[4];
    const float* Wg1  = (const float*)d_in[5];
    const float* bg1  = (const float*)d_in[6];
    const float* Wg2  = (const float*)d_in[7];
    const float* bg2  = (const float*)d_in[8];
    // Wc tuple -> d_in[9..20], bc tuple -> d_in[21..32]

    static const int LN[NT] = {14, 4, 3, 2, 3, 2, 4, 7, 4, 3, 6, 2};

    // workspace carve-up (~106 MB total)
    char* wsb = (char*)d_ws;
    size_t off = 0;
    auto take = [&](size_t bytes) -> char* {
        char* p = wsb + off;
        off += (bytes + 255) & ~(size_t)255;
        return p;
    };
    _Float16* xh     = (_Float16*)take((size_t)Bsz * Dd * 2);        // x in f16
    _Float16* We1T   = (_Float16*)take((size_t)NE * Dd * H1 * 2);    // (E,H1,D)
    _Float16* We2T   = (_Float16*)take((size_t)NE * H1 * H2 * 2);    // (E,H2,H1)
    _Float16* Wg1T   = (_Float16*)take((size_t)NT * Dd * GH * 2);    // (T,GH,D)
    _Float16* hbuf   = (_Float16*)take((size_t)Bsz * H1 * 2);        // reused per expert
    _Float16* ebuf   = (_Float16*)take((size_t)NE * Bsz * H2 * 2);   // experts
    _Float16* gbuf   = (_Float16*)take((size_t)Bsz * GH * 2);        // reused per task
    float*    gatesb = (float*)take((size_t)NT * Bsz * NE * 4);      // gates

    // ---- precision conversion / weight transposition ----
    cvt_f32_to_f16<<<4096, 256, 0, stream>>>(x, xh, (size_t)Bsz * Dd);
    transpose_cvt<<<2048, 256, 0, stream>>>(We1, We1T, NE, Dd, H1);
    transpose_cvt<<<1024, 256, 0, stream>>>(We2, We2T, NE, H1, H2);
    transpose_cvt<<<1024, 256, 0, stream>>>(Wg1, Wg1T, NT, Dd, GH);

    // ---- expert MLPs (per-expert to reuse hidden buffer) ----
    for (int e = 0; e < NE; ++e) {
        dim3 g1(Bsz / TM, H1 / TN);
        gemm_relu_f16<<<g1, 128, 0, stream>>>(
            xh, Dd, We1T + (size_t)e * Dd * H1, be1 + (size_t)e * H1,
            hbuf, H1, Dd);
        dim3 g2(Bsz / TM, H2 / TN);
        gemm_relu_f16<<<g2, 128, 0, stream>>>(
            hbuf, H1, We2T + (size_t)e * H1 * H2, be2 + (size_t)e * H2,
            ebuf + (size_t)e * Bsz * H2, H2, H1);
    }

    // ---- gates (per-task) ----
    for (int t = 0; t < NT; ++t) {
        dim3 gg(Bsz / TM, GH / TN);
        gemm_relu_f16<<<gg, 128, 0, stream>>>(
            xh, Dd, Wg1T + (size_t)t * Dd * GH, bg1 + (size_t)t * GH,
            gbuf, GH, Dd);
        gate2_softmax_kernel<<<Bsz / 256, 256, 0, stream>>>(
            gbuf, Wg2 + (size_t)t * GH * NE, bg2 + (size_t)t * NE,
            gatesb + (size_t)t * Bsz * NE);
    }

    // ---- combine + per-task classifier heads ----
    HeadParams hp;
    size_t ooff = 0;
    float* outp = (float*)d_out;
    for (int i = 0; i < NT; ++i) {
        hp.Wc[i]  = (const float*)d_in[9 + i];
        hp.bc[i]  = (const float*)d_in[21 + i];
        hp.out[i] = outp + ooff;
        hp.n[i]   = LN[i];
        ooff += (size_t)Bsz * LN[i];
    }
    combine_head_kernel<<<dim3(Bsz, NT), 128, 0, stream>>>(ebuf, gatesb, hp);
}
